// HoloAttention_19301583029005
// MI455X (gfx1250) — matrix operand order repaired and checked by reading, not compiled, not run on hardware
//
#include <hip/hip_runtime.h>
#include <hip/hip_bf16.h>
#include <math.h>

typedef __attribute__((ext_vector_type(16))) _Float16 v16h;
typedef __attribute__((ext_vector_type(8)))  float    v8f;
typedef __attribute__((ext_vector_type(4)))  unsigned int u32x4;
typedef __attribute__((ext_vector_type(8)))  int      i32x8;
typedef __attribute__((ext_vector_type(4)))  int      i32x4;

#define BATCH 2
#define SEQ   2048
#define CH    1024
#define NHEAD 16
#define HDIM  64
#define HARM  32
#define BL    (BATCH*SEQ)        // 4096 rows

// ---------------------------------------------------------------------------
// TDM: async 2D-tile load (tile_dim0=64 f16 elems, tile_dim1=32 rows, line
// stride 1024 elems) from global into LDS.  D# packing per CDNA5 ISA 8.3/8.4.
// ---------------------------------------------------------------------------
__device__ __forceinline__ void tdm_load_tile_64x32_f16(const _Float16* gptr,
                                                        unsigned lds_off) {
    unsigned long long ga = (unsigned long long)(uintptr_t)gptr;
    u32x4 g0;
    g0[0] = 1u;                                            // count=1, user D#
    g0[1] = lds_off;                                       // lds_addr (bytes)
    g0[2] = (unsigned)(ga & 0xFFFFFFFFu);                  // global_addr[31:0]
    g0[3] = (unsigned)((ga >> 32) & 0x01FFFFFFu) | (2u << 30); // [56:32] | type=2
    i32x8 g1;
    g1[0] = (1 << 16);        // workgroup_mask=0, data_size=1 (2 bytes)
    g1[1] = (64 << 16);       // tensor_dim0[15:0]=64 at bits 63:48
    g1[2] = (int)(2048u << 16); // tensor_dim1[15:0]=2048 at bits 95:80
    g1[3] = (64 << 16);       // tile_dim0=64 at bits 127:112
    g1[4] = 32;               // tile_dim1=32 at bits 143:128
    g1[5] = 1024;             // tensor_dim0_stride=1024 elems (bits 191:160)
    g1[6] = 0;
    g1[7] = 0;
    i32x4 z4 = {0, 0, 0, 0};
#if __has_include(<hip/amd_detail/amd_gfx1250_TDM.h>)
    i32x8 z8 = {0, 0, 0, 0, 0, 0, 0, 0};
    __builtin_amdgcn_tensor_load_to_lds(g0, g1, z4, z4, z8, 0);
#else
    __builtin_amdgcn_tensor_load_to_lds(g0, g1, z4, z4, 0);
#endif
}

// ---------------------------------------------------------------------------
// 1) f32 -> f16 convert
// ---------------------------------------------------------------------------
__global__ void holo_cvt_f16(const float* __restrict__ src, _Float16* __restrict__ dst, int n) {
    int i = blockIdx.x * blockDim.x + threadIdx.x;
    if (i < n) dst[i] = (_Float16)src[i];
}

// ---------------------------------------------------------------------------
// 2) Effective weight (transposed): Wt[o][i] = sum_h basis[h,i]*amp[o,h]*cos(phase[o,h])
// ---------------------------------------------------------------------------
__global__ void holo_build_weff(const float* __restrict__ basis,
                                const float* __restrict__ phase,
                                const float* __restrict__ amp,
                                _Float16* __restrict__ wt) {
    int idx = blockIdx.x * blockDim.x + threadIdx.x;   // over CH*CH
    int o = idx >> 10;
    int i = idx & (CH - 1);
    float acc = 0.f;
#pragma unroll 8
    for (int h = 0; h < HARM; ++h) {
        float w = amp[o * HARM + h] * __cosf(phase[o * HARM + h]);
        acc = __builtin_fmaf(basis[h * CH + i], w, acc);
    }
    wt[idx] = (_Float16)acc;
}

// ---------------------------------------------------------------------------
// 3) WMMA GEMM:  out[M,N] = alpha * A[M,K] @ Bt[N,K]^T
//    One wave -> 16x64 tile, K stepped by 32 (v_wmma_f32_16x16x32_f16).
// ---------------------------------------------------------------------------
__global__ void holo_gemm_wmma(const _Float16* __restrict__ A,
                               const _Float16* __restrict__ Bt,
                               void* __restrict__ out,
                               int M, int N, int K,
                               float alpha, int out_is_f32) {
    const int lane = threadIdx.x & 31;
    const int g    = lane >> 4;       // half-wave group
    const int m    = lane & 15;
    const int wid  = (blockIdx.x * blockDim.x + threadIdx.x) >> 5;
    const int mtiles = M >> 4;
    const int mt = wid % mtiles;
    const int nt = wid / mtiles;
    if (nt >= (N >> 6)) return;
    const int row0 = mt << 4;
    const int col0 = nt << 6;

    v8f acc[4] = {v8f{}, v8f{}, v8f{}, v8f{}};

    for (int k0 = 0; k0 < K; k0 += 32) {
        // ---- A fragment (16x32 f16): lane m = row, K contiguous (k = 16*g + s) ----
        v16h a;
        const _Float16* ap = A + (size_t)(row0 + m) * K + k0 + g * 16;
        __builtin_prefetch(ap + 32, 0, 1);     // global_prefetch_b8: next k-block
#pragma unroll
        for (int r = 0; r < 8; ++r) {
            const int kb = 2 * r;
            a[2 * r]     = ap[kb];
            a[2 * r + 1] = ap[kb + 1];
        }
#pragma unroll
        for (int n = 0; n < 4; ++n) {
            // ---- B fragment (32x16 f16): lane m = column, K pairs contiguous ----
            v16h b;
            const _Float16* bp = Bt + (size_t)(col0 + n * 16 + m) * K + k0 + g * 16;
#pragma unroll
            for (int r = 0; r < 8; ++r) {
                b[2 * r]     = bp[2 * r];
                b[2 * r + 1] = bp[2 * r + 1];
            }
            acc[n] = __builtin_amdgcn_wmma_f32_16x16x32_f16(
                false, a, false, b, (short)0, acc[n], false, false);
        }
    }

    // ---- store: D layout row = r + 8*g, col = m ----
#pragma unroll
    for (int n = 0; n < 4; ++n) {
#pragma unroll
        for (int r = 0; r < 8; ++r) {
            const int row = row0 + r + g * 8;
            const int col = col0 + n * 16 + m;
            const float v = acc[n][r] * alpha;
            if (out_is_f32) ((float*)out)[(size_t)row * N + col] = v;
            else            ((_Float16*)out)[(size_t)row * N + col] = (_Float16)v;
        }
    }
}

// ---------------------------------------------------------------------------
// 4) Flash attention, TDM-staged K/V.
//    Block = 4 waves on one (b,h), 64 queries; K/V chunks (32 keys x 64d f16)
//    double-buffered in LDS via tensor_load_to_lds, waited with TENSORcnt.
// ---------------------------------------------------------------------------
__global__ void holo_attn_wmma(const _Float16* __restrict__ qb,
                               const _Float16* __restrict__ kb,
                               const _Float16* __restrict__ vb,
                               _Float16* __restrict__ ob) {
    __shared__ _Float16 kt[2][32][64];       // [buf][key][d]   4KB each
    __shared__ _Float16 vt[2][32][64];
    __shared__ _Float16 pshare[4][16][34];   // per-wave P transpose tile

    const int lane  = threadIdx.x & 31;
    const int g     = lane >> 4;
    const int m     = lane & 15;
    const int wslot = threadIdx.x >> 5;      // wave in block (0..3)

    const int qb4 = blockIdx.x & 31;         // 32 query blocks of 64
    const int h   = (blockIdx.x >> 5) & 15;
    const int b   = blockIdx.x >> 9;
    const int q0  = qb4 * 64 + wslot * 16;
    const size_t rowbase = (size_t)b * SEQ;
    const int hc = h * HDIM;

    // ---- load Q fragments (d = 0..31 and 32..63) once ----
    v16h qa[2];
    const _Float16* qrow = qb + (rowbase + q0 + m) * CH + hc;
#pragma unroll
    for (int ks = 0; ks < 2; ++ks) {
#pragma unroll
        for (int r = 0; r < 8; ++r) {
            const int kb_ = 2 * r + g * 16;
            qa[ks][2 * r]     = qrow[ks * 32 + kb_];
            qa[ks][2 * r + 1] = qrow[ks * 32 + kb_ + 1];
        }
    }

    float mr[8], lr[8];
    v8f oacc[4] = {v8f{}, v8f{}, v8f{}, v8f{}};
#pragma unroll
    for (int r = 0; r < 8; ++r) { mr[r] = -1e30f; lr[r] = 0.f; }

    const int NCHUNK = SEQ / 32;             // 64

    // prologue: async-load chunk 0 into buffer 0
    if (wslot == 0) {
        const _Float16* kg = kb + rowbase * CH + hc;
        const _Float16* vg = vb + rowbase * CH + hc;
        tdm_load_tile_64x32_f16(kg, (unsigned)(uintptr_t)&kt[0][0][0]);
        tdm_load_tile_64x32_f16(vg, (unsigned)(uintptr_t)&vt[0][0][0]);
    }

    for (int c = 0; c < NCHUNK; ++c) {
        const int buf = c & 1;
        if (wslot == 0) {
            if (c + 1 < NCHUNK) {   // prefetch next chunk into other buffer
                const size_t nk = rowbase + (size_t)(c + 1) * 32;
                tdm_load_tile_64x32_f16(kb + nk * CH + hc,
                                        (unsigned)(uintptr_t)&kt[buf ^ 1][0][0]);
                tdm_load_tile_64x32_f16(vb + nk * CH + hc,
                                        (unsigned)(uintptr_t)&vt[buf ^ 1][0][0]);
                __builtin_amdgcn_s_wait_tensorcnt(2);   // chunk c landed
            } else {
                __builtin_amdgcn_s_wait_tensorcnt(0);
            }
        }
        __syncthreads();                                 // publish chunk c

        // ---- S = Q @ K^T for 32 keys -> two 16x16 f32 fragments ----
        v8f s[2] = {v8f{}, v8f{}};
#pragma unroll
        for (int nt = 0; nt < 2; ++nt) {
#pragma unroll
            for (int ks = 0; ks < 2; ++ks) {
                v16h bf;
                const _Float16* krow = &kt[buf][nt * 16 + m][ks * 32 + g * 16];
#pragma unroll
                for (int r = 0; r < 8; ++r) {
                    bf[2 * r]     = krow[2 * r];
                    bf[2 * r + 1] = krow[2 * r + 1];
                }
                s[nt] = __builtin_amdgcn_wmma_f32_16x16x32_f16(
                    false, qa[ks], false, bf, (short)0, s[nt], false, false);
            }
        }

        // ---- online softmax: row stats live at (acc reg r, half-wave g) ----
#pragma unroll
        for (int r = 0; r < 8; ++r) {
            float rowmax = fmaxf(s[0][r], s[1][r]);
#pragma unroll
            for (int off = 8; off; off >>= 1)
                rowmax = fmaxf(rowmax, __shfl_xor(rowmax, off, 32));
            const float mnew = fmaxf(mr[r], rowmax);
            const float corr = __expf(mr[r] - mnew);
            mr[r] = mnew;
            const float p0 = __expf(s[0][r] - mnew);
            const float p1 = __expf(s[1][r] - mnew);
            float rs = p0 + p1;
#pragma unroll
            for (int off = 8; off; off >>= 1)
                rs += __shfl_xor(rs, off, 32);
            lr[r] = lr[r] * corr + rs;
#pragma unroll
            for (int nt = 0; nt < 4; ++nt) oacc[nt][r] *= corr;
            pshare[wslot][r + g * 8][m]      = (_Float16)p0;   // D-layout stage
            pshare[wslot][r + g * 8][m + 16] = (_Float16)p1;
        }
        __syncthreads();

        // ---- reload P as A fragment (16 queries x 32 keys) ----
        v16h pa;
#pragma unroll
        for (int r = 0; r < 8; ++r) {
        const int kl = 2 * r + g * 16;
            pa[2 * r]     = pshare[wslot][m][kl];
            pa[2 * r + 1] = pshare[wslot][m][kl + 1];
        }

        // ---- O += P @ V (V B-fragment from LDS) ----
#pragma unroll
        for (int nt = 0; nt < 4; ++nt) {
            v16h vf;
            const int col = nt * 16 + m;
#pragma unroll
            for (int r = 0; r < 8; ++r) {
                vf[2 * r]     = vt[buf][g * 16 + 2 * r][col];
                vf[2 * r + 1] = vt[buf][g * 16 + 2 * r + 1][col];
            }
            oacc[nt] = __builtin_amdgcn_wmma_f32_16x16x32_f16(
                false, pa, false, vf, (short)0, oacc[nt], false, false);
        }
        __syncthreads();     // all reads of buf done before it is overwritten
    }

    // ---- normalize and store [B,L,C] f16 ----
#pragma unroll
    for (int r = 0; r < 8; ++r) {
        const float inv = 1.0f / lr[r];
        const size_t row = rowbase + q0 + r + g * 8;
#pragma unroll
        for (int nt = 0; nt < 4; ++nt)
            ob[row * CH + hc + nt * 16 + m] = (_Float16)(oacc[nt][r] * inv);
    }
}

// ---------------------------------------------------------------------------
// Launcher
// ---------------------------------------------------------------------------
extern "C" void kernel_launch(void* const* d_in, const int* in_sizes, int n_in,
                              void* d_out, int out_size, void* d_ws, size_t ws_size,
                              hipStream_t stream) {
    const float* x  = (const float*)d_in[0];
    const float* bq = (const float*)d_in[1];
    const float* pq = (const float*)d_in[2];
    const float* aq = (const float*)d_in[3];
    const float* bk = (const float*)d_in[4];
    const float* pk = (const float*)d_in[5];
    const float* ak = (const float*)d_in[6];
    const float* bv = (const float*)d_in[7];
    const float* pv = (const float*)d_in[8];
    const float* av = (const float*)d_in[9];
    const float* bo = (const float*)d_in[10];
    const float* po = (const float*)d_in[11];
    const float* ao = (const float*)d_in[12];

    char* ws = (char*)d_ws;
    _Float16* xh   = (_Float16*)(ws);                                    // 8 MB
    _Float16* wq   = (_Float16*)(ws + (size_t)8  * 1024 * 1024);         // 2 MB each
    _Float16* wk   = (_Float16*)(ws + (size_t)10 * 1024 * 1024);
    _Float16* wv   = (_Float16*)(ws + (size_t)12 * 1024 * 1024);
    _Float16* wo   = (_Float16*)(ws + (size_t)14 * 1024 * 1024);
    _Float16* qbuf = (_Float16*)(ws + (size_t)16 * 1024 * 1024);         // 8 MB each
    _Float16* kbuf = (_Float16*)(ws + (size_t)24 * 1024 * 1024);
    _Float16* vbuf = (_Float16*)(ws + (size_t)32 * 1024 * 1024);
    _Float16* abuf = (_Float16*)(ws + (size_t)40 * 1024 * 1024);

    const int nX = BL * CH;                                  // 4194304
    hipLaunchKernelGGL(holo_cvt_f16, dim3((nX + 255) / 256), dim3(256), 0, stream, x, xh, nX);

    const int nW = CH * CH;                                  // 1048576
    hipLaunchKernelGGL(holo_build_weff, dim3(nW / 256), dim3(256), 0, stream, bq, pq, aq, wq);
    hipLaunchKernelGGL(holo_build_weff, dim3(nW / 256), dim3(256), 0, stream, bk, pk, ak, wk);
    hipLaunchKernelGGL(holo_build_weff, dim3(nW / 256), dim3(256), 0, stream, bv, pv, av, wv);
    hipLaunchKernelGGL(holo_build_weff, dim3(nW / 256), dim3(256), 0, stream, bo, po, ao, wo);

    // GEMM grid: (BL/16) * (CH/64) = 4096 waves, 4 waves (128 threads) per block
    const dim3 ggrid(1024), gblk(128);
    const float qscale = 0.125f;   // 1/sqrt(64) folded into q
    hipLaunchKernelGGL(holo_gemm_wmma, ggrid, gblk, 0, stream, xh, wq, (void*)qbuf, BL, CH, CH, qscale, 0);
    hipLaunchKernelGGL(holo_gemm_wmma, ggrid, gblk, 0, stream, xh, wk, (void*)kbuf, BL, CH, CH, 1.0f, 0);
    hipLaunchKernelGGL(holo_gemm_wmma, ggrid, gblk, 0, stream, xh, wv, (void*)vbuf, BL, CH, CH, 1.0f, 0);

    // Attention: B*NH*(L/64) = 1024 blocks x 128 threads (4 waves cooperate)
    hipLaunchKernelGGL(holo_attn_wmma, dim3(1024), dim3(128), 0, stream, qbuf, kbuf, vbuf, abuf);

    // Output projection -> f32 d_out
    hipLaunchKernelGGL(holo_gemm_wmma, ggrid, gblk, 0, stream, abuf, wo, d_out, BL, CH, CH, 1.0f, 1);
}
